// Block_67113158967413
// MI455X (gfx1250) — matrix operand order repaired and checked by reading, not compiled, not run on hardware
//
#include <hip/hip_runtime.h>
#include <math.h>

#define B_   128
#define N_   144
#define D_   768
#define H_   12
#define HD_  64
#define E_   8
#define S_   2
#define DF_  3072
#define NT_  (B_ * N_)     // 18432 tokens
#define SL_  (E_ * S_)     // 16 slots
#define QKVW (3 * D_)      // 2304
#define VP_  160           // padded token dim for transposed V / P tiles

typedef __bf16 bf16;
typedef __attribute__((ext_vector_type(16))) __bf16 bf16x16;
typedef __attribute__((ext_vector_type(8)))  float  floatx8;
typedef __attribute__((__vector_size__(16))) int    v4i_vec;

// ---- CDNA5 async global->LDS staging (guarded; falls back to plain copy) ----
#if defined(__has_builtin)
#if __has_builtin(__builtin_amdgcn_global_load_async_to_lds_b128)
#define HAVE_ASYNC_LDS 1
#endif
#endif

#ifdef HAVE_ASYNC_LDS
// Signature (from hipcc diagnostic): (global int4*, lds int4*, imm, cpol)
#define ASYNC_CP(gp, lp)                                                      \
  __builtin_amdgcn_global_load_async_to_lds_b128(                             \
      (__attribute__((address_space(1))) v4i_vec*)(gp),                       \
      (__attribute__((address_space(3))) v4i_vec*)(lp), 0, 0)
#if __has_builtin(__builtin_amdgcn_s_wait_asynccnt)
#define ASYNC_WAIT() __builtin_amdgcn_s_wait_asynccnt(0)
#else
#define ASYNC_WAIT() asm volatile("s_wait_asynccnt 0" ::: "memory")
#endif
#else
#define ASYNC_CP(gp, lp) (*(uint4*)(lp) = *(const uint4*)(gp))
#define ASYNC_WAIT() ((void)0)
#endif

union FragU { uint4 q[2]; bf16x16 v; };

// A-fragment (16x32 bf16): lane holds row = lane&15,
// elements 0..15 = K = kb..kb+15, kb = (lane>=16)*16
__device__ __forceinline__ bf16x16 load_a_frag_gap(const bf16* rowbase, int k0) {
  int lane = threadIdx.x & 31;
  int kb = (lane >> 4) << 4;
  const bf16* p = rowbase + k0 + kb;
  FragU u;
  u.q[0] = *(const uint4*)(p);
  u.q[1] = *(const uint4*)(p + 8);
  return u.v;
}

// 16 contiguous bf16: B-fragment when the 16 K-values of one column are
// contiguous (transposed weights / K rows / transposed V / LDS tiles)
__device__ __forceinline__ bf16x16 load_frag_contig(const bf16* base) {
  FragU u;
  u.q[0] = *(const uint4*)(base);
  u.q[1] = *(const uint4*)(base + 8);
  return u.v;
}

// ---------------------------------------------------------------------------
// Weight convert + transpose: dst[z][n][k] = (bf16) src[z][k][n]
// ---------------------------------------------------------------------------
__global__ __launch_bounds__(256)
void transpose_conv_kernel(const float* __restrict__ src, bf16* __restrict__ dst,
                           int K, int Nn) {
  __shared__ float tile[32][33];
  int z = blockIdx.z;
  const float* s = src + (size_t)z * K * Nn;
  bf16* d = dst + (size_t)z * K * Nn;
  int k0 = blockIdx.y * 32, n0 = blockIdx.x * 32;
  int tx = threadIdx.x & 31, ty = threadIdx.x >> 5;  // 32 x 8
#pragma unroll
  for (int i = 0; i < 32; i += 8)
    tile[ty + i][tx] = s[(size_t)(k0 + ty + i) * Nn + (n0 + tx)];
  __syncthreads();
#pragma unroll
  for (int i = 0; i < 32; i += 8)
    d[(size_t)(n0 + ty + i) * K + (k0 + tx)] = (bf16)tile[tx][ty + i];
}

// ---------------------------------------------------------------------------
// LayerNorm (fp32 in) -> bf16 out
// ---------------------------------------------------------------------------
__global__ __launch_bounds__(256)
void ln_kernel(const float* __restrict__ x, const float* __restrict__ w,
               const float* __restrict__ b, bf16* __restrict__ out) {
  __shared__ float red[2][8];
  int t = blockIdx.x;
  const float* xr = x + (size_t)t * D_;
  float v[3];
  float s = 0.f, sq = 0.f;
#pragma unroll
  for (int i = 0; i < 3; ++i) {
    v[i] = xr[threadIdx.x + i * 256];
    s += v[i];
    sq += v[i] * v[i];
  }
#pragma unroll
  for (int m = 16; m >= 1; m >>= 1) { s += __shfl_xor(s, m); sq += __shfl_xor(sq, m); }
  int wave = threadIdx.x >> 5, lane = threadIdx.x & 31;
  if (lane == 0) { red[0][wave] = s; red[1][wave] = sq; }
  __syncthreads();
  float ts = 0.f, tsq = 0.f;
#pragma unroll
  for (int i = 0; i < 8; ++i) { ts += red[0][i]; tsq += red[1][i]; }
  float mean = ts * (1.f / D_);
  float var = tsq * (1.f / D_) - mean * mean;
  float rs = rsqrtf(var + 1e-5f);
  bf16* orow = out + (size_t)t * D_;
#pragma unroll
  for (int i = 0; i < 3; ++i) {
    int d = threadIdx.x + i * 256;
    orow[d] = (bf16)((v[i] - mean) * rs * w[d] + b[d]);
  }
}

// ---------------------------------------------------------------------------
// WMMA GEMM: C = A(bf16)[M,K] @ B[K,N] with B given TRANSPOSED bf16 Bt[N,K].
// 8 waves/block, block tile 256x64, wave tile 32x64 (2x4 acc tiles).
// Per k-step each wave: 2 A-frags (global, reused x4) + 4 B-frags (LDS,
// preloaded, reused x2) -> 8 WMMAs. B tile (64x32 = 4 KB) staged in LDS via
// async global->LDS copy, double-buffered, shared by all 8 waves.
// moe!=0 -> expert-batched row map (blockIdx.z = e):
// arow = (row>>1)*16 + e*2 + (row&1)   (xs/mid/ys stored [B, 16, *]).
// ---------------------------------------------------------------------------
__device__ __forceinline__ int map_row(int r, int moe, int e) {
  return moe ? ((r >> 1) * SL_ + e * S_ + (r & 1)) : r;
}

__global__ __launch_bounds__(256)
void wmma_gemm_kernel(const bf16* __restrict__ A, const bf16* __restrict__ Bt,
                      const float* __restrict__ bias, int bias_zstride,
                      const float* __restrict__ resid,
                      float* __restrict__ outF, bf16* __restrict__ outB,
                      int Nn, int K, int moe, int do_gelu) {
  __shared__ bf16 Bs[2][64][32];
  int lane = threadIdx.x & 31;
  int wave = threadIdx.x >> 5;
  int half = lane >> 4;
  int e = blockIdx.z;
  int m0 = blockIdx.y * 256 + wave * 32;
  int n0 = blockIdx.x * 64;

  // staging map: thread -> (B column sr, 8-element chunk sc)
  int sr = threadIdx.x >> 2;
  int sc = (threadIdx.x & 3) << 3;
  const bf16* gB = Bt + (size_t)e * K * Nn + (size_t)(n0 + sr) * K + sc;

  const bf16* arow0 = A + (size_t)map_row(m0 + (lane & 15), moe, e) * K;
  const bf16* arow1 = A + (size_t)map_row(m0 + 16 + (lane & 15), moe, e) * K;

  floatx8 acc[2][4] = {};

  ASYNC_CP(gB, &Bs[0][sr][sc]);
  ASYNC_WAIT();
  __syncthreads();

  int buf = 0;
  for (int k0 = 0; k0 < K; k0 += 32) {
    int nxt = buf ^ 1;
    if (k0 + 32 < K) ASYNC_CP(gB + k0 + 32, &Bs[nxt][sr][sc]);

    bf16x16 a0 = load_a_frag_gap(arow0, k0);
    bf16x16 a1 = load_a_frag_gap(arow1, k0);
    bf16x16 bfr[4];
#pragma unroll
    for (int j = 0; j < 4; ++j)
      bfr[j] = load_frag_contig(&Bs[buf][j * 16 + (lane & 15)][(lane >> 4) << 4]);
#pragma unroll
    for (int j = 0; j < 4; ++j) {
      acc[0][j] = __builtin_amdgcn_wmma_f32_16x16x32_bf16(false, a0, false, bfr[j],
                                                          (short)0, acc[0][j], false, false);
      acc[1][j] = __builtin_amdgcn_wmma_f32_16x16x32_bf16(false, a1, false, bfr[j],
                                                          (short)0, acc[1][j], false, false);
    }
    ASYNC_WAIT();
    __syncthreads();
    buf = nxt;
  }

#pragma unroll
  for (int i = 0; i < 2; ++i) {
#pragma unroll
    for (int j = 0; j < 4; ++j) {
      int col = n0 + j * 16 + (lane & 15);
      float bv = bias ? bias[(size_t)e * bias_zstride + col] : 0.f;
#pragma unroll
      for (int r = 0; r < 8; ++r) {
        int rr = m0 + i * 16 + r + half * 8;
        int orow = map_row(rr, moe, e);
        float v = acc[i][j][r] + bv;
        if (do_gelu) v = 0.5f * v * (1.f + erff(v * 0.70710678118f));
        if (resid) v += resid[(size_t)orow * Nn + col];
        if (outF) outF[(size_t)orow * Nn + col] = v;
        else      outB[(size_t)orow * Nn + col] = (bf16)v;
      }
    }
  }
}

// ---------------------------------------------------------------------------
// V transpose for attention: vt[(b*H+h)*64 + d][n] = V[b,n,h,d], n padded->160
// ---------------------------------------------------------------------------
__global__ __launch_bounds__(256)
void vtrans_kernel(const bf16* __restrict__ qkv, bf16* __restrict__ vt) {
  int bh = blockIdx.x;
  int b = bh / H_, h = bh % H_;
  const bf16* vsrc = qkv + (size_t)(b * N_) * QKVW + 2 * D_ + h * HD_;
  bf16* vd = vt + (size_t)bh * HD_ * VP_;
  for (int idx = threadIdx.x; idx < HD_ * VP_; idx += 256) {
    int d = idx / VP_, n = idx % VP_;
    vd[idx] = (n < N_) ? vsrc[(size_t)n * QKVW + d] : (bf16)0.f;
  }
}

// ---------------------------------------------------------------------------
// Attention: one wave per (16-row q-tile, head, batch).
// ---------------------------------------------------------------------------
__global__ __launch_bounds__(32)
void attn_kernel(const bf16* __restrict__ qkv, const bf16* __restrict__ vt,
                 bf16* __restrict__ o) {
  __shared__ bf16 P[16][VP_];  // probabilities, K padded 144 -> 160
  int lane = threadIdx.x & 31;
  int half = lane >> 4;
  int qt = blockIdx.x * 16;
  int h = blockIdx.y;
  int b = blockIdx.z;

  const bf16* base = qkv + (size_t)(b * N_) * QKVW + h * HD_;

  // ---- scores S = Q K^T : 16 x 144 in 9 accumulator tiles ----
  floatx8 sc[9] = {};
  const bf16* qrow = base + (size_t)(qt + (lane & 15)) * QKVW;
#pragma unroll
  for (int k0 = 0; k0 < HD_; k0 += 32) {
    bf16x16 a = load_a_frag_gap(qrow, k0);
#pragma unroll
    for (int t = 0; t < 9; ++t) {
      int j = t * 16 + (lane & 15);  // key index (B column)
      const bf16* kp = base + D_ + (size_t)j * QKVW + k0 + ((lane >> 4) << 4);
      bf16x16 bfr = load_frag_contig(kp);
      sc[t] = __builtin_amdgcn_wmma_f32_16x16x32_bf16(false, a, false, bfr,
                                                      (short)0, sc[t], false, false);
    }
  }

  // ---- row softmax (row in VGPR r + 8*half; 16 cols across lanes of a half) --
  const float scale = 0.125f;  // hd^-0.5
#pragma unroll
  for (int r = 0; r < 8; ++r) {
    float m = -1e30f;
#pragma unroll
    for (int t = 0; t < 9; ++t) m = fmaxf(m, sc[t][r]);
    m = fmaxf(m, __shfl_xor(m, 1));
    m = fmaxf(m, __shfl_xor(m, 2));
    m = fmaxf(m, __shfl_xor(m, 4));
    m = fmaxf(m, __shfl_xor(m, 8));
    m *= scale;
    float s = 0.f;
#pragma unroll
    for (int t = 0; t < 9; ++t) {
      float p = __expf(sc[t][r] * scale - m);
      sc[t][r] = p;
      s += p;
    }
    s += __shfl_xor(s, 1);
    s += __shfl_xor(s, 2);
    s += __shfl_xor(s, 4);
    s += __shfl_xor(s, 8);
    float inv = 1.f / s;
#pragma unroll
    for (int t = 0; t < 9; ++t)
      P[r + half * 8][t * 16 + (lane & 15)] = (bf16)(sc[t][r] * inv);
  }
  for (int idx = lane; idx < 16 * 16; idx += 32)
    P[idx >> 4][144 + (idx & 15)] = (bf16)0.f;
  __syncthreads();

  // ---- O = P V : 16 x 64, V read from transposed/padded vt ----
  const bf16* vbase = vt + (size_t)(b * H_ + h) * HD_ * VP_;
  floatx8 oacc[4] = {};
  for (int k0 = 0; k0 < VP_; k0 += 32) {
    bf16x16 a = load_a_frag_gap(&P[lane & 15][0], k0);
#pragma unroll
    for (int j = 0; j < 4; ++j) {
      int cold = j * 16 + (lane & 15);
      const bf16* vp = vbase + (size_t)cold * VP_ + k0 + ((lane >> 4) << 4);
      bf16x16 bfr = load_frag_contig(vp);
      oacc[j] = __builtin_amdgcn_wmma_f32_16x16x32_bf16(false, a, false, bfr,
                                                        (short)0, oacc[j], false, false);
    }
  }
#pragma unroll
  for (int j = 0; j < 4; ++j) {
    int col = h * HD_ + j * 16 + (lane & 15);
#pragma unroll
    for (int r = 0; r < 8; ++r) {
      int n = qt + r + half * 8;
      o[(size_t)(b * N_ + n) * D_ + col] = (bf16)oacc[j][r];
    }
  }
}

// ---------------------------------------------------------------------------
// logits = h2 @ phi, dispatch = softmax over 16 slots (per token)
// ---------------------------------------------------------------------------
__global__ __launch_bounds__(128)
void logits_kernel(const bf16* __restrict__ h2, const float* __restrict__ phi,
                   float* __restrict__ logits, float* __restrict__ disp) {
  __shared__ float sl[SL_ + 2];
  int t = blockIdx.x;
  int s = threadIdx.x >> 3, j = threadIdx.x & 7;
  const bf16* hr = h2 + (size_t)t * D_;
  float acc = 0.f;
  for (int d = j; d < D_; d += 8) acc += (float)hr[d] * phi[d * SL_ + s];
  acc += __shfl_xor(acc, 1);
  acc += __shfl_xor(acc, 2);
  acc += __shfl_xor(acc, 4);
  if (j == 0) sl[s] = acc;
  __syncthreads();
  if (threadIdx.x == 0) {
    float m = -1e30f;
    for (int i = 0; i < SL_; ++i) m = fmaxf(m, sl[i]);
    float su = 0.f;
    for (int i = 0; i < SL_; ++i) su += __expf(sl[i] - m);
    sl[SL_] = m;
    sl[SL_ + 1] = su;
  }
  __syncthreads();
  if (threadIdx.x < SL_) {
    float l = sl[threadIdx.x];
    logits[(size_t)t * SL_ + threadIdx.x] = l;
    disp[(size_t)t * SL_ + threadIdx.x] = __expf(l - sl[SL_]) / sl[SL_ + 1];
  }
}

// combine = softmax over tokens (axis=1) per (b, slot)
__global__ __launch_bounds__(256)
void combine_kernel(const float* __restrict__ logits, float* __restrict__ comb) {
  __shared__ float red[8];
  int s = blockIdx.x, b = blockIdx.y;
  int n = threadIdx.x;
  int wave = threadIdx.x >> 5, lane = threadIdx.x & 31;
  float l = (n < N_) ? logits[(size_t)(b * N_ + n) * SL_ + s] : -1e30f;
  float m = l;
#pragma unroll
  for (int mm = 16; mm >= 1; mm >>= 1) m = fmaxf(m, __shfl_xor(m, mm));
  if (lane == 0) red[wave] = m;
  __syncthreads();
  float tm = -1e30f;
#pragma unroll
  for (int i = 0; i < 8; ++i) tm = fmaxf(tm, red[i]);
  __syncthreads();
  float e = (n < N_) ? __expf(l - tm) : 0.f;
  float su = e;
#pragma unroll
  for (int mm = 16; mm >= 1; mm >>= 1) su += __shfl_xor(su, mm);
  if (lane == 0) red[wave] = su;
  __syncthreads();
  float ts = 0.f;
#pragma unroll
  for (int i = 0; i < 8; ++i) ts += red[i];
  if (n < N_) comb[(size_t)(b * N_ + n) * SL_ + s] = e / ts;
}

// xs[b, slot, :] = sum_n dispatch[b,n,slot] * h2[b,n,:]
__global__ __launch_bounds__(256)
void xs_kernel(const float* __restrict__ disp, const bf16* __restrict__ h2,
               bf16* __restrict__ xs) {
  __shared__ float dw[N_];
  int s = blockIdx.x, b = blockIdx.y;
  if (threadIdx.x < N_) dw[threadIdx.x] = disp[(size_t)(b * N_ + threadIdx.x) * SL_ + s];
  __syncthreads();
#pragma unroll
  for (int i = 0; i < 3; ++i) {
    int d = threadIdx.x + i * 256;
    float acc = 0.f;
    for (int n = 0; n < N_; ++n)
      acc += dw[n] * (float)h2[(size_t)(b * N_ + n) * D_ + d];
    xs[(size_t)(b * SL_ + s) * D_ + d] = (bf16)acc;
  }
}

// out = x1 + combine @ ys
__global__ __launch_bounds__(256)
void final_kernel(const float* __restrict__ x1, const float* __restrict__ comb,
                  const float* __restrict__ ys, float* __restrict__ out) {
  __shared__ float cw[SL_];
  int t = blockIdx.x;
  int b = t / N_;
  if (threadIdx.x < SL_) cw[threadIdx.x] = comb[(size_t)t * SL_ + threadIdx.x];
  __syncthreads();
#pragma unroll
  for (int i = 0; i < 3; ++i) {
    int d = threadIdx.x + i * 256;
    float acc = x1[(size_t)t * D_ + d];
#pragma unroll
    for (int s = 0; s < SL_; ++s)
      acc += cw[s] * ys[(size_t)(b * SL_ + s) * D_ + d];
    out[(size_t)t * D_ + d] = acc;
  }
}

// ---------------------------------------------------------------------------
extern "C" void kernel_launch(void* const* d_in, const int* in_sizes, int n_in,
                              void* d_out, int out_size, void* d_ws, size_t ws_size,
                              hipStream_t stream) {
  (void)in_sizes; (void)n_in; (void)out_size; (void)ws_size;
  const float* x      = (const float*)d_in[0];
  const float* ln1_w  = (const float*)d_in[1];
  const float* ln1_b  = (const float*)d_in[2];
  const float* ln2_w  = (const float*)d_in[3];
  const float* ln2_b  = (const float*)d_in[4];
  const float* qkv_w  = (const float*)d_in[5];
  const float* proj_w = (const float*)d_in[6];
  const float* proj_b = (const float*)d_in[7];
  const float* phi    = (const float*)d_in[8];
  const float* w1     = (const float*)d_in[9];
  const float* b1     = (const float*)d_in[10];
  const float* w2     = (const float*)d_in[11];
  const float* b2     = (const float*)d_in[12];

  char* ws = (char*)d_ws;
  size_t off = 0;
  auto alloc = [&](size_t bytes) -> char* {
    char* p = ws + off;
    off += (bytes + 255) & ~(size_t)255;
    return p;
  };

  bf16*  h      = (bf16*)alloc((size_t)NT_ * D_ * 2);
  bf16*  qkvb   = (bf16*)alloc((size_t)NT_ * QKVW * 2);
  bf16*  ob     = (bf16*)alloc((size_t)NT_ * D_ * 2);
  float* x1     = (float*)alloc((size_t)NT_ * D_ * 4);
  bf16*  h2     = (bf16*)alloc((size_t)NT_ * D_ * 2);
  float* logits = (float*)alloc((size_t)NT_ * SL_ * 4);
  float* disp   = (float*)alloc((size_t)NT_ * SL_ * 4);
  float* comb   = (float*)alloc((size_t)NT_ * SL_ * 4);
  bf16*  xsb    = (bf16*)alloc((size_t)B_ * SL_ * D_ * 2);
  bf16*  midb   = (bf16*)alloc((size_t)B_ * SL_ * DF_ * 2);
  float* ysf    = (float*)alloc((size_t)B_ * SL_ * D_ * 4);
  bf16*  qkvt   = (bf16*)alloc((size_t)D_ * QKVW * 2);          // [2304][768]
  bf16*  projt  = (bf16*)alloc((size_t)D_ * D_ * 2);            // [768][768]
  bf16*  w1t    = (bf16*)alloc((size_t)E_ * D_ * DF_ * 2);      // [8][3072][768]
  bf16*  w2t    = (bf16*)alloc((size_t)E_ * DF_ * D_ * 2);      // [8][768][3072]
  bf16*  vt     = (bf16*)alloc((size_t)B_ * H_ * HD_ * VP_ * 2);

  // 0) one-time-per-call weight convert+transpose (bf16, [N][K])
  transpose_conv_kernel<<<dim3(QKVW / 32, D_ / 32, 1), 256, 0, stream>>>(qkv_w, qkvt, D_, QKVW);
  transpose_conv_kernel<<<dim3(D_ / 32, D_ / 32, 1), 256, 0, stream>>>(proj_w, projt, D_, D_);
  transpose_conv_kernel<<<dim3(DF_ / 32, D_ / 32, E_), 256, 0, stream>>>(w1, w1t, D_, DF_);
  transpose_conv_kernel<<<dim3(D_ / 32, DF_ / 32, E_), 256, 0, stream>>>(w2, w2t, DF_, D_);

  // 1) LN1 -> h
  ln_kernel<<<NT_, 256, 0, stream>>>(x, ln1_w, ln1_b, h);
  // 2) qkv = h @ qkv_w
  wmma_gemm_kernel<<<dim3(QKVW / 64, NT_ / 256, 1), 256, 0, stream>>>(
      h, qkvt, nullptr, 0, nullptr, nullptr, qkvb, QKVW, D_, 0, 0);
  // 3) transposed V copy, then attention
  vtrans_kernel<<<B_ * H_, 256, 0, stream>>>(qkvb, vt);
  attn_kernel<<<dim3(N_ / 16, H_, B_), 32, 0, stream>>>(qkvb, vt, ob);
  // 4) x1 = x + o @ proj_w + proj_b
  wmma_gemm_kernel<<<dim3(D_ / 64, NT_ / 256, 1), 256, 0, stream>>>(
      ob, projt, proj_b, 0, x, x1, nullptr, D_, D_, 0, 0);
  // 5) LN2 -> h2
  ln_kernel<<<NT_, 256, 0, stream>>>(x1, ln2_w, ln2_b, h2);
  // 6) logits + dispatch softmax
  logits_kernel<<<NT_, 128, 0, stream>>>(h2, phi, logits, disp);
  // 7) combine softmax
  combine_kernel<<<dim3(SL_, B_), 256, 0, stream>>>(logits, comb);
  // 8) xs = dispatch^T @ h2
  xs_kernel<<<dim3(SL_, B_), 256, 0, stream>>>(disp, h2, xsb);
  // 9) mid = gelu(xs @ w1 + b1), expert-batched (M = 256 rows -> grid.y = 1)
  wmma_gemm_kernel<<<dim3(DF_ / 64, (B_ * S_) / 256, E_), 256, 0, stream>>>(
      xsb, w1t, b1, DF_, nullptr, nullptr, midb, DF_, D_, 1, 1);
  // 10) ys = mid @ w2 + b2, expert-batched
  wmma_gemm_kernel<<<dim3(D_ / 64, (B_ * S_) / 256, E_), 256, 0, stream>>>(
      midb, w2t, b2, D_, nullptr, ysf, nullptr, D_, DF_, 1, 0);
  // 11) out = x1 + combine @ ys
  final_kernel<<<NT_, 256, 0, stream>>>(x1, comb, ysf, (float*)d_out);
}
